// ResBlock_60584808677761
// MI455X (gfx1250) — matrix operand order repaired and checked by reading, not compiled, not run on hardware
//
#include <hip/hip_runtime.h>

typedef __attribute__((ext_vector_type(16))) __bf16 v16bf;
typedef __attribute__((ext_vector_type(8)))  float  v8f;
typedef __attribute__((ext_vector_type(4)))  unsigned int v4u;
typedef __attribute__((ext_vector_type(2)))  unsigned int v2u;

union Frag { v16bf v; v4u q[2]; };

__device__ __forceinline__ unsigned short f2bf(float f) {
    unsigned u = __float_as_uint(f);
    u += 0x7FFFu + ((u >> 16) & 1u);        // round-to-nearest-even
    return (unsigned short)(u >> 16);
}

// ---------------- weight pack: OIHW f32 -> [co][tap*256+ci] bf16 ----------------
__global__ __launch_bounds__(256) void pack_w(const float* __restrict__ w,
                                              unsigned short* __restrict__ wp) {
    int idx = blockIdx.x * 256 + threadIdx.x;          // 256*2304 = 589824
    int co  = idx / 2304;
    int k   = idx - co * 2304;
    int tap = k >> 8;                                  // kh*3+kw
    int ci  = k & 255;
    wp[idx] = f2bf(w[(co * 256 + ci) * 9 + tap]);
}

// ---------------- mask: 1x1 conv -> hard binary ----------------
__global__ __launch_bounds__(256) void mask_kernel(const float* __restrict__ x,
                                                   const float* __restrict__ wm,
                                                   const float* __restrict__ b,
                                                   float* __restrict__ mask) {
    __shared__ float swm[256];
    swm[threadIdx.x] = wm[threadIdx.x];
    __syncthreads();
    int pix = blockIdx.x * 256 + threadIdx.x;          // 16*64*64 = 65536
    int n   = pix >> 12;
    int hw  = pix & 4095;
    const float* xp = x + (size_t)n * 256 * 4096 + hw;
    float s = b[0];
    #pragma unroll 8
    for (int ci = 0; ci < 256; ++ci) s += xp[(size_t)ci * 4096] * swm[ci];
    mask[pix] = (s > 0.f) ? 1.f : 0.f;
}

// ---------------- 3x3 max-pool dilation of binary mask ----------------
__global__ __launch_bounds__(256) void dilate_kernel(const float* __restrict__ m,
                                                     float* __restrict__ md) {
    int pix = blockIdx.x * 256 + threadIdx.x;
    int n = pix >> 12, h = (pix >> 6) & 63, w = pix & 63;
    const float* mn = m + ((size_t)n << 12);
    float v = 0.f;
    for (int dh = -1; dh <= 1; ++dh) {
        int hh = h + dh; if ((unsigned)hh > 63u) continue;
        for (int dw = -1; dw <= 1; ++dw) {
            int ww = w + dw; if ((unsigned)ww > 63u) continue;
            float t = mn[hh * 64 + ww];
            v = v > t ? v : t;
        }
    }
    md[pix] = v;
}

// ---------------- 3x3 conv as implicit GEMM via bf16 WMMA ----------------
// MODE 1: out = relu(conv(in)*gate)                 (h1, gate = dilated mask)
// MODE 2: out = resid + relu(conv(in)*gate)*gate    (layer2, gate = mask)
#define CIP 72   // padded ci stride in LDS (bank-conflict-free b128/b64 access)

template<int MODE>
__global__ __launch_bounds__(256) void conv_wmma(
    const float* __restrict__ in,             // [16,256,64,64]
    const unsigned short* __restrict__ wp,    // [256][2304] bf16 packed
    const float* __restrict__ gate,           // [16,64,64]
    const float* __restrict__ resid,          // x (MODE 2)
    float* __restrict__ out)                  // [16,256,64,64]
{
    __shared__ unsigned short slab[3 * 66 * CIP];      // 28.5 KB
    const int tid  = threadIdx.x;
    const int h    = blockIdx.x;
    const int n    = blockIdx.y;
    const int wave = tid >> 5;                 // wave32
    const int lane = tid & 31;
    const int lrow = lane & 15;
    const int hi   = lane >> 4;                // 0|1 half-wave

    v8f acc[2][4] = {};

    // zero the w=-1 / w=64 padding columns (persist across ci-chunks)
    for (int e = tid; e < 3 * 2 * 64; e += 256) {
        int r = e / 128, rem = e - r * 128;
        int side = rem >> 6, ci = rem & 63;
        slab[(r * 66 + (side ? 65 : 0)) * CIP + ci] = 0;
    }

    // -------- software-pipelined slab staging --------
    // Each thread owns 12 packed elements: e = (row, ci-group-of-4, w).
    float rbuf[12][4];

    auto load_chunk = [&](int cc) {
        #pragma unroll
        for (int i = 0; i < 12; ++i) {
            int e  = tid + i * 256;            // < 3*16*64 = 3072
            int r  = e >> 10;
            int rm = e & 1023;
            int c4 = rm >> 6;
            int w  = rm & 63;
            int hr = h + r - 1;
            if ((unsigned)hr < 64u) {
                const float* p = in + (((size_t)n * 256 + cc * 64 + c4 * 4) << 12)
                                    + (hr << 6) + w;
                rbuf[i][0] = p[0];
                rbuf[i][1] = p[4096];
                rbuf[i][2] = p[8192];
                rbuf[i][3] = p[12288];
            } else {
                rbuf[i][0] = rbuf[i][1] = rbuf[i][2] = rbuf[i][3] = 0.f;
            }
        }
    };
    auto store_chunk = [&]() {
        #pragma unroll
        for (int i = 0; i < 12; ++i) {
            int e  = tid + i * 256;
            int r  = e >> 10;
            int rm = e & 1023;
            int c4 = rm >> 6;
            int w  = rm & 63;
            unsigned lo = (unsigned)f2bf(rbuf[i][0]) | ((unsigned)f2bf(rbuf[i][1]) << 16);
            unsigned hh = (unsigned)f2bf(rbuf[i][2]) | ((unsigned)f2bf(rbuf[i][3]) << 16);
            v2u* dst = reinterpret_cast<v2u*>(slab + (r * 66 + w + 1) * CIP + c4 * 4);
            v2u pk; pk.x = lo; pk.y = hh;
            *dst = pk;                          // ds_store_b64, conflict-free
        }
    };

    const int m0base = wave * 32;

    load_chunk(0);
    for (int cc = 0; cc < 4; ++cc) {                   // 64-ci chunks
        __syncthreads();                               // slab free (prev compute done)
        store_chunk();
        __syncthreads();
        if (cc < 3) load_chunk(cc + 1);                // global loads fly during WMMA

        for (int t = 0; t < 9; ++t) {                  // taps kh*3+kw
            const int row = t / 3, kw = t - row * 3;
            #pragma unroll
            for (int s = 0; s < 2; ++s) {              // 32-K sub-chunks
                const int Kbase = t * 256 + cc * 64 + s * 32;
                Frag a[2], b[4];
                #pragma unroll
                for (int mt = 0; mt < 2; ++mt) {       // A: lane holds K{0..7,16..23}/{8..15,24..31}
                    const v4u* ap = reinterpret_cast<const v4u*>(
                        wp + (size_t)(m0base + mt * 16 + lrow) * 2304 + Kbase + hi * 8);
                    a[mt].q[0] = ap[0];
                    a[mt].q[1] = ap[2];
                }
                #pragma unroll
                for (int nt = 0; nt < 4; ++nt) {       // B: lane = col, K{0..7,16..23}/{8..15,24..31}
                    const v4u* bp = reinterpret_cast<const v4u*>(
                        slab + (row * 66 + nt * 16 + lrow + kw) * CIP + s * 32 + hi * 8);
                    b[nt].q[0] = bp[0];
                    b[nt].q[1] = bp[2];
                }
                #pragma unroll
                for (int mt = 0; mt < 2; ++mt)
                    #pragma unroll
                    for (int nt = 0; nt < 4; ++nt)
                        acc[mt][nt] = __builtin_amdgcn_wmma_f32_16x16x32_bf16(
                            false, a[mt].v, false, b[nt].v,
                            (short)0, acc[mt][nt], false, false);
            }
        }
    }

    // epilogue: gate * relu (+ residual + second gate for MODE 2)
    const int gbase = (n << 12) + (h << 6);
    #pragma unroll
    for (int nt = 0; nt < 4; ++nt) {
        const int w = nt * 16 + lrow;
        const float g = gate[gbase + w];
        #pragma unroll
        for (int mt = 0; mt < 2; ++mt) {
            const int mco = m0base + mt * 16 + hi * 8;
            #pragma unroll
            for (int v = 0; v < 8; ++v) {
                const size_t oidx = (((size_t)n * 256 + mco + v) << 12) + (h << 6) + w;
                float val = acc[mt][nt][v] * g;
                val = val > 0.f ? val : 0.f;
                if (MODE == 2) out[oidx] = resid[oidx] + val * g;
                else           out[oidx] = val;
            }
        }
    }
}

extern "C" void kernel_launch(void* const* d_in, const int* in_sizes, int n_in,
                              void* d_out, int out_size, void* d_ws, size_t ws_size,
                              hipStream_t stream) {
    const float* x  = (const float*)d_in[0];   // [16,256,64,64]
    const float* w1 = (const float*)d_in[1];   // [256,256,3,3]
    const float* w2 = (const float*)d_in[2];
    const float* wm = (const float*)d_in[3];   // [1,256,1,1]
    const float* bm = (const float*)d_in[4];   // [1]
    float* out = (float*)d_out;
    char* ws = (char*)d_ws;

    unsigned short* w1p  = (unsigned short*)(ws);               // 1,179,648 B
    unsigned short* w2p  = (unsigned short*)(ws + 1179648);     // 1,179,648 B
    float*          mask = (float*)(ws + 2359296);              //   262,144 B
    float*          mkd  = (float*)(ws + 2621440);              //   262,144 B
    float*          hbuf = (float*)(ws + 2883584);              // 67,108,864 B

    pack_w<<<2304, 256, 0, stream>>>(w1, w1p);
    pack_w<<<2304, 256, 0, stream>>>(w2, w2p);
    mask_kernel<<<256, 256, 0, stream>>>(x, wm, bm, mask);
    dilate_kernel<<<256, 256, 0, stream>>>(mask, mkd);

    dim3 grid(64, 16);  // (h, n)
    conv_wmma<1><<<grid, 256, 0, stream>>>(x,    w1p, mkd,  nullptr, hbuf);
    conv_wmma<2><<<grid, 256, 0, stream>>>(hbuf, w2p, mask, x,       out);
}
